// MultiHeadSelfAttention_4956392259833
// MI455X (gfx1250) — matrix operand order repaired and checked by reading, not compiled, not run on hardware
//
#include <hip/hip_runtime.h>
#include <hip/hip_bf16.h>
#include <stdint.h>

// Problem constants (match reference)
#define EMB   1024
#define NHEAD 16
#define HDIM  64
#define BATCH 4
#define SEQ   2048
#define MTOT  (BATCH*SEQ)   // 8192 rows

typedef __bf16 bf16_t;
typedef bf16_t v16bf __attribute__((ext_vector_type(16)));
typedef float  v8f   __attribute__((ext_vector_type(8)));

union Frag {
    v16bf v;
    uint4 q[2];
};

__device__ __forceinline__ unsigned short f2bf(float f) {
    unsigned u = __float_as_uint(f);
    unsigned r = (u + 0x7FFFu + ((u >> 16) & 1u)) >> 16;   // RNE truncate
    return (unsigned short)r;
}

__device__ __forceinline__ v8f wmma_bf16(const Frag& a, const Frag& b, v8f c) {
    // D = A(16x32 bf16) x B(32x16 bf16) + C(16x16 f32)
    return __builtin_amdgcn_wmma_f32_16x16x32_bf16(
        false, a.v, false, b.v, (short)0, c, false, false);
}

// ---------------------------------------------------------------------------
// Async global -> LDS copy (16 B per lane), ASYNCcnt-tracked.
// Builtin expects (int4 AS1*, int4 AS3*, imm offset, imm cpol).
// ---------------------------------------------------------------------------
typedef int i32x4 __attribute__((vector_size(16)));
typedef __attribute__((address_space(1))) i32x4* as1_i32x4_p;
typedef __attribute__((address_space(3))) i32x4* as3_i32x4_p;

__device__ __forceinline__ void async_load_b128(const void* g, void* l) {
#if __has_builtin(__builtin_amdgcn_global_load_async_to_lds_b128)
    __builtin_amdgcn_global_load_async_to_lds_b128(
        (as1_i32x4_p)(unsigned long long)(size_t)g,
        (as3_i32x4_p)(unsigned)(size_t)l,
        0, 0);
#else
    unsigned lofs = (unsigned)(size_t)l;
    unsigned long long ga = (unsigned long long)(size_t)g;
    asm volatile("global_load_async_to_lds_b128 %0, %1, off"
                 :: "v"(lofs), "v"(ga) : "memory");
#endif
}

template<int N>
__device__ __forceinline__ void wait_asynccnt() {
#if __has_builtin(__builtin_amdgcn_s_wait_asynccnt)
    __builtin_amdgcn_s_wait_asynccnt((short)N);
#else
    asm volatile("s_wait_asynccnt %0" :: "i"(N) : "memory");
#endif
}

// ---------------------------------------------------------------------------
// 1) fp32 -> bf16 cast (vectorized x4)
// ---------------------------------------------------------------------------
__global__ void cvt_f32_bf16(const float* __restrict__ src,
                             unsigned short* __restrict__ dst, int n4) {
    int i = blockIdx.x * blockDim.x + threadIdx.x;
    if (i < n4) {
        float4 f = ((const float4*)src)[i];
        ushort4 o;
        o.x = f2bf(f.x); o.y = f2bf(f.y); o.z = f2bf(f.z); o.w = f2bf(f.w);
        ((ushort4*)dst)[i] = o;
    }
}

// ---------------------------------------------------------------------------
// 2) NT GEMM via WMMA with async double-buffered LDS staging.
//    C[M,N] = A[M,K] (row-major) * W[N,K]^T   (nn.Linear convention)
//    Block: 256 thr = 8 waves; block tile 128x128; wave tile 64x32.
//    Per k-step (32): stage 128x32 A-tile + 128x32 B-tile into LDS via
//    GLOBAL_LOAD_ASYNC_TO_LDS_B128 (ASYNCcnt), overlap with WMMA on the
//    other buffer; s_wait_asynccnt 4 fences only the current tile.
//    LDS rows padded to 40 elements (80 B) to spread banks.
// ---------------------------------------------------------------------------
#define LDSTRIDE 40   // elements per LDS tile row (32 data + 8 pad)

__device__ __forceinline__ void stage_tile(const unsigned short* __restrict__ gbase,
                                           int K, int k0,
                                           unsigned short* lbase, int tid) {
    // 128 rows x 32 elems = 512 chunks of 16 B; 2 chunks per thread.
#pragma unroll
    for (int cc = 0; cc < 2; ++cc) {
        int c   = tid + cc * 256;
        int row = c >> 2;
        int ce  = (c & 3) * 8;               // element offset within row
        async_load_b128(gbase + (size_t)row * K + k0 + ce,
                        lbase + row * LDSTRIDE + ce);
    }
}

template<bool OUTF32>
__global__ void __launch_bounds__(256, 1)
gemm_nt(const unsigned short* __restrict__ A,
        const unsigned short* __restrict__ W,
        void* __restrict__ Cout,
        const float* __restrict__ bias,
        int M, int N, int K) {
    __shared__ unsigned short lA[2][128 * LDSTRIDE];
    __shared__ unsigned short lB[2][128 * LDSTRIDE];

    const int tid  = threadIdx.x;
    const int lane = tid & 31;
    const int hl   = lane >> 4;        // half-wave
    const int r    = lane & 15;
    const int wave = tid >> 5;
    const int wm   = wave & 1;
    const int wn   = wave >> 1;
    const int bm   = blockIdx.x * 128;
    const int bn   = blockIdx.y * 128;

    const unsigned short* Ab = A + (size_t)bm * K;
    const unsigned short* Wb = W + (size_t)bn * K;

    v8f acc[4][2];
    v8f vzero = {0.f,0.f,0.f,0.f,0.f,0.f,0.f,0.f};
#pragma unroll
    for (int i = 0; i < 4; ++i)
#pragma unroll
        for (int j = 0; j < 2; ++j) acc[i][j] = vzero;

    // Prologue: stage tile 0 into buffer 0 (4 async ops per thread).
    stage_tile(Ab, K, 0, &lA[0][0], tid);
    stage_tile(Wb, K, 0, &lB[0][0], tid);

    const int nk = K >> 5;
    for (int i = 0; i < nk; ++i) {
        const int cur = i & 1;
        if (i + 1 < nk) {
            // Stage next tile into the other buffer, then fence only the
            // 4 outstanding transfers of the *current* tile.
            stage_tile(Ab, K, (i + 1) << 5, &lA[cur ^ 1][0], tid);
            stage_tile(Wb, K, (i + 1) << 5, &lB[cur ^ 1][0], tid);
            wait_asynccnt<4>();
        } else {
            wait_asynccnt<0>();
        }
        __syncthreads();                 // current tile visible to all waves

        // Fragment reads from LDS (ds_load_b128) + 8 WMMAs.
        Frag af[4];
#pragma unroll
        for (int mt = 0; mt < 4; ++mt) {
            const unsigned short* p =
                &lA[cur][(wm*64 + mt*16 + r) * LDSTRIDE + hl*8];
            af[mt].q[0] = *(const uint4*)p;          // K {0..7 | 8..15}
            af[mt].q[1] = *(const uint4*)(p + 16);   // K {16..23 | 24..31}
        }
        Frag bfg[2];
#pragma unroll
        for (int nt = 0; nt < 2; ++nt) {
            const unsigned short* p =
                &lB[cur][(wn*32 + nt*16 + r) * LDSTRIDE + hl*8];
            bfg[nt].q[0] = *(const uint4*)p;         // K {0..7 | 8..15}
            bfg[nt].q[1] = *(const uint4*)(p + 16);  // K {16..23 | 24..31}
        }
#pragma unroll
        for (int mt = 0; mt < 4; ++mt)
#pragma unroll
            for (int nt = 0; nt < 2; ++nt)
                acc[mt][nt] = wmma_bf16(af[mt], bfg[nt], acc[mt][nt]);

        __syncthreads();                 // reads done before buffer reuse
    }

#pragma unroll
    for (int mt = 0; mt < 4; ++mt)
#pragma unroll
        for (int nt = 0; nt < 2; ++nt)
#pragma unroll
            for (int j = 0; j < 8; ++j) {
                int row = bm + wm*64 + mt*16 + hl*8 + j;  // C: M = j + 8*half
                int col = bn + wn*32 + nt*16 + r;         // N = lane%16
                float vv = acc[mt][nt][j];
                if (OUTF32) {
                    ((float*)Cout)[(size_t)row * N + col] = vv + bias[col];
                } else {
                    ((unsigned short*)Cout)[(size_t)row * N + col] = f2bf(vv);
                }
            }
}

// ---------------------------------------------------------------------------
// 3) V transpose to [b, h, d, n] so PV B-fragments are contiguous over keys
// ---------------------------------------------------------------------------
__global__ void transpose_v(const unsigned short* __restrict__ V,
                            unsigned short* __restrict__ Vt) {
    int i = blockIdx.x * blockDim.x + threadIdx.x;   // over MTOT*EMB elements
    int e   = i & (EMB - 1);
    int row = i >> 10;                               // EMB = 1024
    int h = e >> 6, d = e & 63;
    int b = row >> 11, n = row & (SEQ - 1);          // SEQ = 2048
    Vt[(((size_t)(b*NHEAD + h)*HDIM + d) << 11) + n] = V[i];
}

// ---------------------------------------------------------------------------
// 4) Flash attention forward. Each wave: one 16-row query tile of one (b,h).
//    Key tiles of 32: 4 WMMAs for S (16x32), online softmax (lane-xor
//    reductions within each 16-lane half), P staged through per-wave LDS to
//    build the A-fragment, 4 WMMAs for O += P * V.
// ---------------------------------------------------------------------------
__global__ void __launch_bounds__(256, 1)
attn_fwd(const unsigned short* __restrict__ Qb,
         const unsigned short* __restrict__ Kb,
         const unsigned short* __restrict__ Vt,
         unsigned short* __restrict__ Ctx) {
    __shared__ unsigned short plds[8][16 * 40];      // 16x32 P tile, stride 40

    const int lane = threadIdx.x & 31;
    const int hl   = lane >> 4;
    const int r    = lane & 15;
    const int wave = threadIdx.x >> 5;
    const int bh   = blockIdx.y;                     // b*NHEAD + h
    const int b    = bh >> 4;
    const int h    = bh & 15;
    const int q0   = blockIdx.x * 128 + wave * 16;

    const float scale = 0.125f;                      // 1/sqrt(64)

    // Q A-fragments: 16 rows x 64 dims -> 2 frags (d-chunks 0..31, 32..63)
    Frag qf[2];
    {
        const unsigned short* qp =
            Qb + (size_t)(b*SEQ + q0 + r) * EMB + h*HDIM;
#pragma unroll
        for (int t = 0; t < 2; ++t) {
            const unsigned short* p = qp + t*32 + hl*8;
            qf[t].q[0] = *(const uint4*)p;
            qf[t].q[1] = *(const uint4*)(p + 16);
        }
    }

    v8f vzero = {0.f,0.f,0.f,0.f,0.f,0.f,0.f,0.f};
    v8f o[4];
#pragma unroll
    for (int t = 0; t < 4; ++t) o[t] = vzero;
    float mrow[8], lrow[8];
#pragma unroll
    for (int j = 0; j < 8; ++j) { mrow[j] = -3.0e38f; lrow[j] = 0.f; }

    for (int kt = 0; kt < SEQ; kt += 32) {
        // K B-fragments: B = K^T (d x key); lane = key column, contiguous d.
        Frag kf[2][2];                               // [key subtile][d chunk]
#pragma unroll
        for (int s = 0; s < 2; ++s) {
            const unsigned short* kp =
                Kb + (size_t)(b*SEQ + kt + s*16 + r) * EMB + h*HDIM + hl*8;
#pragma unroll
            for (int t = 0; t < 2; ++t) {
                kf[s][t].q[0] = *(const uint4*)(kp + t*32);
                kf[s][t].q[1] = *(const uint4*)(kp + t*32 + 16);
            }
        }
        v8f s0 = vzero, s1 = vzero;
        s0 = wmma_bf16(qf[0], kf[0][0], s0);
        s0 = wmma_bf16(qf[1], kf[0][1], s0);
        s1 = wmma_bf16(qf[0], kf[1][0], s1);
        s1 = wmma_bf16(qf[1], kf[1][1], s1);

        // Online softmax across the 32 keys of this tile.
#pragma unroll
        for (int j = 0; j < 8; ++j) {
            float a0 = s0[j] * scale, a1 = s1[j] * scale;
            float mx = fmaxf(a0, a1);
#pragma unroll
            for (int off = 1; off < 16; off <<= 1)
                mx = fmaxf(mx, __shfl_xor(mx, off, 32));
            float mnew = fmaxf(mrow[j], mx);
            float p0 = __expf(a0 - mnew);
            float p1 = __expf(a1 - mnew);
            float rs = p0 + p1;
#pragma unroll
            for (int off = 1; off < 16; off <<= 1)
                rs += __shfl_xor(rs, off, 32);
            float alpha = __expf(mrow[j] - mnew);
            lrow[j] = lrow[j] * alpha + rs;
            mrow[j] = mnew;
            o[0][j] *= alpha; o[1][j] *= alpha;
            o[2][j] *= alpha; o[3][j] *= alpha;
            int prow = hl*8 + j;                     // C layout row
            plds[wave][prow*40 + r]      = f2bf(p0); // keys 0..15
            plds[wave][prow*40 + 16 + r] = f2bf(p1); // keys 16..31
        }
        // Cross-lane LDS RAW: make P stores visible before A-frag reads.
        asm volatile("s_wait_dscnt 0x0" ::: "memory");

        Frag pa;                                     // P as A-fragment 16x32
        {
            const unsigned short* p = &plds[wave][r*40 + hl*8];
            pa.q[0] = *(const uint4*)p;
            pa.q[1] = *(const uint4*)(p + 16);
        }

        // V B-fragments from Vt[b,h,d,n]: lane = d column, contiguous keys.
#pragma unroll
        for (int t = 0; t < 4; ++t) {
            Frag vf;
            const unsigned short* p =
                Vt + (size_t)(bh*HDIM + t*16 + r) * SEQ + kt + hl*8;
            vf.q[0] = *(const uint4*)p;
            vf.q[1] = *(const uint4*)(p + 16);
            o[t] = wmma_bf16(pa, vf, o[t]);
        }
    }

    // Normalize and emit context in [b, n, h*64+d] layout (bf16).
#pragma unroll
    for (int j = 0; j < 8; ++j) {
        float inv = 1.0f / lrow[j];
        int row = q0 + hl*8 + j;
        unsigned short* cp = Ctx + (size_t)(b*SEQ + row) * EMB + h*HDIM;
#pragma unroll
        for (int t = 0; t < 4; ++t) cp[t*16 + r] = f2bf(o[t][j] * inv);
    }
}

// ---------------------------------------------------------------------------
// Launcher
// ---------------------------------------------------------------------------
extern "C" void kernel_launch(void* const* d_in, const int* in_sizes, int n_in,
                              void* d_out, int out_size, void* d_ws, size_t ws_size,
                              hipStream_t stream) {
    const float* z  = (const float*)d_in[0];
    const float* wq = (const float*)d_in[1];
    const float* wk = (const float*)d_in[2];
    const float* wv = (const float*)d_in[3];
    const float* wo = (const float*)d_in[4];
    const float* bo = (const float*)d_in[5];
    float* out = (float*)d_out;

    char* ws = (char*)d_ws;
    // bf16 workspace layout (bytes):
    unsigned short* zb  = (unsigned short*)(ws + 0);          // 16 MiB
    unsigned short* wqb = (unsigned short*)(ws + 16777216);   //  2 MiB
    unsigned short* wkb = (unsigned short*)(ws + 18874368);
    unsigned short* wvb = (unsigned short*)(ws + 20971520);
    unsigned short* wob = (unsigned short*)(ws + 23068672);
    unsigned short* qb  = (unsigned short*)(ws + 25165824);   // 16 MiB
    unsigned short* kb  = (unsigned short*)(ws + 41943040);   // 16 MiB
    unsigned short* vb  = (unsigned short*)(ws + 58720256);   // 16 MiB
    unsigned short* ctx = (unsigned short*)(ws + 75497472);   // 16 MiB
    unsigned short* vt  = (unsigned short*)(ws + 0);          // reuses zb (dead)

    // 1) casts to bf16
    {
        int n4 = (MTOT * EMB) / 4;
        cvt_f32_bf16<<<(n4 + 255) / 256, 256, 0, stream>>>(z, zb, n4);
        n4 = (EMB * EMB) / 4;
        cvt_f32_bf16<<<(n4 + 255) / 256, 256, 0, stream>>>(wq, wqb, n4);
        cvt_f32_bf16<<<(n4 + 255) / 256, 256, 0, stream>>>(wk, wkb, n4);
        cvt_f32_bf16<<<(n4 + 255) / 256, 256, 0, stream>>>(wv, wvb, n4);
        cvt_f32_bf16<<<(n4 + 255) / 256, 256, 0, stream>>>(wo, wob, n4);
    }

    // 2) QKV projections: [8192,1024] = zb @ W^T
    dim3 gg(MTOT / 128, EMB / 128);   // 64 x 8 blocks
    gemm_nt<false><<<gg, 256, 0, stream>>>(zb, wqb, qb, nullptr, MTOT, EMB, EMB);
    gemm_nt<false><<<gg, 256, 0, stream>>>(zb, wkb, kb, nullptr, MTOT, EMB, EMB);
    gemm_nt<false><<<gg, 256, 0, stream>>>(zb, wvb, vb, nullptr, MTOT, EMB, EMB);

    // 3) V -> Vt [b,h,d,n]   (zb is dead from here; vt aliases it)
    transpose_v<<<(MTOT * EMB) / 256, 256, 0, stream>>>(vb, vt);

    // 4) flash attention: grid = (query tiles of 128, B*H)
    dim3 ga(SEQ / 128, BATCH * NHEAD);  // 16 x 64 blocks
    attn_fwd<<<ga, 256, 0, stream>>>(qb, kb, vt, ctx);

    // 5) output projection + bias, fp32 out
    gemm_nt<true><<<gg, 256, 0, stream>>>(ctx, wob, out, bo, MTOT, EMB, EMB);
}
